// MinecraftAwareAttention3D_84748294685328
// MI455X (gfx1250) — compile-verified, matrix operand order checked by reading, not run on hardware
//
#include <hip/hip_runtime.h>
#include <hip/hip_bf16.h>

typedef __attribute__((ext_vector_type(16))) _Float16 v16h;
typedef __attribute__((ext_vector_type(8)))  float    v8f;

#define N_TOK 4096
#define C_CH  256
#define HD    32
#define NH    8
#define NEG_BIG (-1000000000.0f)

// ---------------------------------------------------------------------------
// Fragment loader: CDNA5 wave32 WMMA f16 A/B layout.
// Lane L holds row/col (L&15); its 16 halves cover K = {8h..8h+7} U {16+8h..16+8h+7}
// where h = L>>4.  Source must be row-major with leading dim `ldh` (halves).
// ---------------------------------------------------------------------------
__device__ __forceinline__ v16h ld_frag(const _Float16* __restrict__ p, int ldh, int lane) {
  const int l16 = lane & 15, half = lane >> 4;
  const _Float16* q = p + (size_t)l16 * ldh + (half << 3);
  v16h r;
#pragma unroll
  for (int j = 0; j < 8; ++j) { r[j] = q[j]; r[j + 8] = q[j + 16]; }
  return r;
}

__device__ __forceinline__ v8f wmma_f16(v16h a, v16h b, v8f c) {
  return __builtin_amdgcn_wmma_f32_16x16x32_f16(false, a, false, b, (short)0, c, false, false);
}

// raw LDS offset (addrspace(3) numeric value) for async-to-LDS VDST operand
__device__ __forceinline__ unsigned lds_addr(void* p) {
  return (unsigned)(size_t)(__attribute__((address_space(3))) void*)p;
}

__device__ __forceinline__ void async_ld_b128(unsigned ldsoff, const void* gaddr) {
  asm volatile("global_load_async_to_lds_b128 %0, %1, off"
               :: "v"(ldsoff), "v"((unsigned long long)(size_t)gaddr)
               : "memory");
}

// ---------------------------------------------------------------------------
// Kernel 1: weight f32->f16 conversion + per-voxel material gather
// ---------------------------------------------------------------------------
__global__ void prep_kernel(const float* __restrict__ qkv_w, const float* __restrict__ proj_w,
                            const int* __restrict__ bt,
                            const float* __restrict__ is_air, const float* __restrict__ is_wood,
                            const float* __restrict__ is_leaves,
                            _Float16* __restrict__ wq, _Float16* __restrict__ wp,
                            float* __restrict__ airN, float* __restrict__ woodN,
                            float* __restrict__ leavesN) {
  int idx = blockIdx.x * blockDim.x + threadIdx.x;
  if (idx < 3 * C_CH * C_CH) wq[idx] = (_Float16)qkv_w[idx];
  if (idx < C_CH * C_CH)     wp[idx] = (_Float16)proj_w[idx];
  if (idx < N_TOK) {
    int b = bt[idx];
    airN[idx]    = is_air[b];
    woodN[idx]   = is_wood[b];
    leavesN[idx] = is_leaves[b];
  }
}

// ---------------------------------------------------------------------------
// Kernel 2: group-norm statistics (8 groups of 32 channels x 4096 voxels)
// ---------------------------------------------------------------------------
__global__ void gn_stats_kernel(const float* __restrict__ x, float* __restrict__ stats) {
  const int g = blockIdx.x, t = threadIdx.x;
  const float* xg = x + (size_t)g * 32 * N_TOK;
  float s = 0.f, ss = 0.f;
  for (int i = t; i < 32 * N_TOK; i += 256) { float v = xg[i]; s += v; ss += v * v; }
  __shared__ float sh_s[256], sh_q[256];
  sh_s[t] = s; sh_q[t] = ss; __syncthreads();
  for (int off = 128; off > 0; off >>= 1) {
    if (t < off) { sh_s[t] += sh_s[t + off]; sh_q[t] += sh_q[t + off]; }
    __syncthreads();
  }
  if (t == 0) {
    const float inv = 1.0f / (32.0f * (float)N_TOK);
    float mu  = sh_s[0] * inv;
    float var = sh_q[0] * inv - mu * mu;
    stats[2 * g]     = mu;
    stats[2 * g + 1] = rsqrtf(var + 1e-5f);
  }
}

// ---------------------------------------------------------------------------
// Kernel 3: apply group norm, write h transposed as f16: hT[n][c]
// ---------------------------------------------------------------------------
__global__ void gn_apply_kernel(const float* __restrict__ x, const float* __restrict__ stats,
                                const float* __restrict__ gw, const float* __restrict__ gb,
                                _Float16* __restrict__ hT) {
  int idx = blockIdx.x * blockDim.x + threadIdx.x;
  if (idx >= C_CH * N_TOK) return;
  int c = idx & (C_CH - 1);
  int n = idx >> 8;
  int g = c >> 5;
  float mu = stats[2 * g], rs = stats[2 * g + 1];
  float v = x[(size_t)c * N_TOK + n];
  hT[idx] = (_Float16)((v - mu) * rs * gw[c] + gb[c]);
}

// ---------------------------------------------------------------------------
// Kernel 4: QKV GEMM (768x256 @ 256x4096) via WMMA f16, fp32 accum.
// Q,K -> [head][n][32] (d contiguous), V -> [head][32][n] (n contiguous).
// ---------------------------------------------------------------------------
__global__ void qkv_gemm_kernel(const _Float16* __restrict__ wq, const _Float16* __restrict__ hT,
                                const float* __restrict__ qkv_b,
                                _Float16* __restrict__ qb, _Float16* __restrict__ kb,
                                _Float16* __restrict__ vb) {
  const int lane = threadIdx.x & 31;
  const int wave = threadIdx.x >> 5;
  const int obase = blockIdx.x * 16;
  const int nbase = (blockIdx.y * 4 + wave) * 16;
  v8f acc = {};
#pragma unroll
  for (int kc = 0; kc < C_CH; kc += 32) {
    v16h a = ld_frag(wq + (size_t)obase * C_CH + kc, C_CH, lane);
    v16h b = ld_frag(hT + (size_t)nbase * C_CH + kc, C_CH, lane);
    acc = wmma_f16(a, b, acc);
  }
  const int l16 = lane & 15, half = lane >> 4;
  const int n = nbase + l16;
#pragma unroll
  for (int r = 0; r < 8; ++r) {
    int o = obase + r + (half << 3);
    float v = acc[r] + qkv_b[o];
    int which = o >> 8;          // 0=q, 1=k, 2=v
    int oc = o & 255;
    int head = oc >> 5, d = oc & 31;
    if (which == 0)      qb[((size_t)head * N_TOK + n) * HD + d] = (_Float16)v;
    else if (which == 1) kb[((size_t)head * N_TOK + n) * HD + d] = (_Float16)v;
    else                 vb[((size_t)head * HD + d) * N_TOK + n] = (_Float16)v;
  }
}

// ---------------------------------------------------------------------------
// Kernel 5: flash attention. K/V panels are DMA'd into double-buffered LDS
// with GLOBAL_LOAD_ASYNC_TO_LDS_B128 (ASYNCcnt), shared by all 4 waves.
// ---------------------------------------------------------------------------
__device__ __forceinline__ void stage_kv(const _Float16* __restrict__ Kh,
                                         const _Float16* __restrict__ Vh,
                                         int m0, _Float16* kst, _Float16* vst,
                                         int wave, int lane) {
  // 8 async b128 instructions move K(32x32 f16 = 2KB) + V(32x32 f16 = 2KB);
  // wave w issues instructions 2w and 2w+1. Each instr: 32 lanes x 16B = 512B.
  const int sub = lane >> 2;          // row within group of 8
  const int q16 = (lane & 3) << 4;    // byte offset within 64B row
#pragma unroll
  for (int t = 0; t < 2; ++t) {
    int j = wave * 2 + t;
    if (j < 4) {                       // K rows (m-local), 64B per row
      int row = j * 8 + sub;
      const char* g = (const char*)(Kh + (size_t)(m0 + row) * HD) + q16;
      async_ld_b128(lds_addr((char*)kst + row * 64 + q16), g);
    } else {                           // V rows (d), 64B per row = this m-chunk
      int d = (j - 4) * 8 + sub;
      const char* g = (const char*)(Vh + (size_t)d * N_TOK + m0) + q16;
      async_ld_b128(lds_addr((char*)vst + d * 64 + q16), g);
    }
  }
}

__global__ void attention_kernel(const _Float16* __restrict__ qbuf, const _Float16* __restrict__ kbuf,
                                 const _Float16* __restrict__ vbuf,
                                 const float* __restrict__ airN, const float* __restrict__ woodN,
                                 const float* __restrict__ leavesN,
                                 _Float16* __restrict__ aT) {
  const int lane = threadIdx.x & 31;
  const int wave = threadIdx.x >> 5;
  const int head = blockIdx.x;
  const int rb   = blockIdx.y * 64 + wave * 16;
  const int l16 = lane & 15, half = lane >> 4;

  __shared__ __align__(16) _Float16 kstage[2][32 * 32];   // [buf][m-local][d]
  __shared__ __align__(16) _Float16 vstage[2][32 * 32];   // [buf][d][m-local]
  __shared__ __align__(16) _Float16 pbuf[4][16 * 32];     // per-wave P staging

  const _Float16* Qh = qbuf + (size_t)head * N_TOK * HD;  // [n][d]
  const _Float16* Kh = kbuf + (size_t)head * N_TOK * HD;  // [m][d]
  const _Float16* Vh = vbuf + (size_t)head * HD * N_TOK;  // [d][m]

  v16h qa = ld_frag(Qh + (size_t)rb * HD, HD, lane);      // persistent A fragment

  float wood_n[8], leaf_n[8], y_n[8];
#pragma unroll
  for (int r = 0; r < 8; ++r) {
    int n = rb + r + (half << 3);
    wood_n[r] = woodN[n];
    leaf_n[r] = leavesN[n];
    y_n[r]    = (float)((n >> 4) & 15);
  }

  v8f o0 = {}, o1 = {};
  float mrun[8], lrun[8];
#pragma unroll
  for (int r = 0; r < 8; ++r) { mrun[r] = -1e30f; lrun[r] = 0.f; }
  const float scale = 0.17677669529663687f;   // 1/sqrt(32)
  const v8f zf = {};

  // prologue: stage chunk 0 into buffer 0
  stage_kv(Kh, Vh, 0, kstage[0], vstage[0], wave, lane);
  asm volatile("s_wait_asynccnt 0x0" ::: "memory");
  __syncthreads();

  for (int m0 = 0; m0 < N_TOK; m0 += 32) {
    const int cur = (m0 >> 5) & 1, nxt = cur ^ 1;
    if (m0 + 32 < N_TOK)            // DMA next panel while we compute this one
      stage_kv(Kh, Vh, m0 + 32, kstage[nxt], vstage[nxt], wave, lane);

    // ---- scores: two 16x16 tiles, fragments read from LDS ----
    v16h kf0 = ld_frag(kstage[cur], 32, lane);            // m-local 0..15
    v16h kf1 = ld_frag(kstage[cur] + 16 * 32, 32, lane);  // m-local 16..31
    v8f s0 = wmma_f16(qa, kf0, zf);
    v8f s1 = wmma_f16(qa, kf1, zf);

    // ---- per-column metadata ----
    const int mc0 = m0 + l16, mc1 = m0 + 16 + l16;
    const bool  a0m = airN[mc0] > 0.f, a1m = airN[mc1] > 0.f;
    const float wd0 = woodN[mc0],   wd1 = woodN[mc1];
    const float lv0 = leavesN[mc0], lv1 = leavesN[mc1];
    const float y0 = (float)((mc0 >> 4) & 15), y1 = (float)((mc1 >> 4) & 15);

    // indicators are 0/1 and mutually exclusive, so all reference clips are
    // no-ops: boost = wood_n*wood_m*(2 + 1.5*vert) + 2*leaf_n*leaf_m
    float p0[8], p1[8], rmx[8];
#pragma unroll
    for (int r = 0; r < 8; ++r) {
      float vm0 = (fabsf(y_n[r] - y0) <= 2.f) ? 1.5f : 0.f;
      float vm1 = (fabsf(y_n[r] - y1) <= 2.f) ? 1.5f : 0.f;
      float base0 = a0m ? NEG_BIG : s0[r] * scale;
      float base1 = a1m ? NEG_BIG : s1[r] * scale;
      float a = base0 + wood_n[r] * wd0 * (2.0f + vm0) + 2.0f * (leaf_n[r] * lv0);
      float b = base1 + wood_n[r] * wd1 * (2.0f + vm1) + 2.0f * (leaf_n[r] * lv1);
      p0[r] = a; p1[r] = b;
      rmx[r] = fmaxf(a, b);
    }

    // ---- row max across the 16 lanes holding one row ----
#pragma unroll
    for (int off = 1; off < 16; off <<= 1)
#pragma unroll
      for (int r = 0; r < 8; ++r) rmx[r] = fmaxf(rmx[r], __shfl_xor(rmx[r], off, 32));

    float rsum[8];
#pragma unroll
    for (int r = 0; r < 8; ++r) {
      float nm   = fmaxf(mrun[r], rmx[r]);
      float corr = __expf(mrun[r] - nm);
      mrun[r] = nm;
      lrun[r] *= corr;
      o0[r] = o0[r] * corr;
      o1[r] = o1[r] * corr;
      float e0 = __expf(p0[r] - nm);
      float e1 = __expf(p1[r] - nm);
      p0[r] = e0; p1[r] = e1;
      rsum[r] = e0 + e1;
    }
#pragma unroll
    for (int off = 1; off < 16; off <<= 1)
#pragma unroll
      for (int r = 0; r < 8; ++r) rsum[r] += __shfl_xor(rsum[r], off, 32);
#pragma unroll
    for (int r = 0; r < 8; ++r) lrun[r] += rsum[r];

    // ---- D-layout P -> A-layout via per-wave LDS tile ----
    _Float16* pb = &pbuf[wave][0];
#pragma unroll
    for (int r = 0; r < 8; ++r) {
      int row = r + (half << 3);
      pb[row * 32 + l16]      = (_Float16)p0[r];
      pb[row * 32 + 16 + l16] = (_Float16)p1[r];
    }
    asm volatile("s_wait_dscnt 0x0" ::: "memory");   // same-wave cross-lane LDS RAW
    v16h pa = ld_frag(pb, 32, lane);

    // ---- O += P @ V^T : V fragments from LDS ----
    v16h vf0 = ld_frag(vstage[cur], 32, lane);            // d = 0..15
    v16h vf1 = ld_frag(vstage[cur] + 16 * 32, 32, lane);  // d = 16..31
    o0 = wmma_f16(pa, vf0, o0);
    o1 = wmma_f16(pa, vf1, o1);

    // drain this wave's DMA + LDS reads, then block-wide buffer handoff
    asm volatile("s_wait_asynccnt 0x0\n\ts_wait_dscnt 0x0" ::: "memory");
    __syncthreads();
  }

  // ---- normalize and store transposed f16: aT[n][head*32+d] ----
#pragma unroll
  for (int r = 0; r < 8; ++r) {
    int n = rb + r + (half << 3);
    float inv = 1.0f / lrun[r];
    aT[(size_t)n * C_CH + head * HD + l16]      = (_Float16)(o0[r] * inv);
    aT[(size_t)n * C_CH + head * HD + 16 + l16] = (_Float16)(o1[r] * inv);
  }
}

// ---------------------------------------------------------------------------
// Kernel 6: output projection via WMMA + bias + residual -> fp32 output
// ---------------------------------------------------------------------------
__global__ void proj_gemm_kernel(const _Float16* __restrict__ wp, const _Float16* __restrict__ aT,
                                 const float* __restrict__ proj_b, const float* __restrict__ x,
                                 float* __restrict__ out) {
  const int lane = threadIdx.x & 31;
  const int wave = threadIdx.x >> 5;
  const int obase = blockIdx.x * 16;
  const int nbase = (blockIdx.y * 4 + wave) * 16;
  v8f acc = {};
#pragma unroll
  for (int kc = 0; kc < C_CH; kc += 32) {
    v16h a = ld_frag(wp + (size_t)obase * C_CH + kc, C_CH, lane);
    v16h b = ld_frag(aT + (size_t)nbase * C_CH + kc, C_CH, lane);
    acc = wmma_f16(a, b, acc);
  }
  const int l16 = lane & 15, half = lane >> 4;
  const int n = nbase + l16;
#pragma unroll
  for (int r = 0; r < 8; ++r) {
    int o = obase + r + (half << 3);
    size_t idx = (size_t)o * N_TOK + n;
    out[idx] = x[idx] + acc[r] + proj_b[o];
  }
}

// ---------------------------------------------------------------------------
extern "C" void kernel_launch(void* const* d_in, const int* in_sizes, int n_in,
                              void* d_out, int out_size, void* d_ws, size_t ws_size,
                              hipStream_t stream) {
  (void)in_sizes; (void)n_in; (void)out_size; (void)ws_size;
  const float* x        = (const float*)d_in[0];
  const int*   bt       = (const int*)  d_in[1];
  const float* gn_w     = (const float*)d_in[2];
  const float* gn_b     = (const float*)d_in[3];
  const float* qkv_w    = (const float*)d_in[4];
  const float* qkv_b    = (const float*)d_in[5];
  const float* proj_w   = (const float*)d_in[6];
  const float* proj_b   = (const float*)d_in[7];
  const float* is_air   = (const float*)d_in[8];
  const float* is_wood  = (const float*)d_in[9];
  const float* is_leaves= (const float*)d_in[10];
  float* out = (float*)d_out;

  char* ws = (char*)d_ws;
  size_t off = 0;
  auto carve = [&](size_t bytes) -> void* {
    void* p = ws + off;
    off = (off + bytes + 255) & ~(size_t)255;
    return p;
  };
  _Float16* hT  = (_Float16*)carve((size_t)N_TOK * C_CH * 2);
  _Float16* wq  = (_Float16*)carve((size_t)3 * C_CH * C_CH * 2);
  _Float16* wp  = (_Float16*)carve((size_t)C_CH * C_CH * 2);
  _Float16* qb  = (_Float16*)carve((size_t)NH * N_TOK * HD * 2);
  _Float16* kb  = (_Float16*)carve((size_t)NH * N_TOK * HD * 2);
  _Float16* vb  = (_Float16*)carve((size_t)NH * HD * N_TOK * 2);
  _Float16* aT  = (_Float16*)carve((size_t)N_TOK * C_CH * 2);
  float* airN    = (float*)carve(N_TOK * 4);
  float* woodN   = (float*)carve(N_TOK * 4);
  float* leavesN = (float*)carve(N_TOK * 4);
  float* stats   = (float*)carve(64);

  prep_kernel<<<(3 * C_CH * C_CH + 255) / 256, 256, 0, stream>>>(
      qkv_w, proj_w, bt, is_air, is_wood, is_leaves, wq, wp, airN, woodN, leavesN);
  gn_stats_kernel<<<8, 256, 0, stream>>>(x, stats);
  gn_apply_kernel<<<(C_CH * N_TOK) / 256, 256, 0, stream>>>(x, stats, gn_w, gn_b, hT);
  qkv_gemm_kernel<<<dim3(48, 64), 128, 0, stream>>>(wq, hT, qkv_b, qb, kb, vb);
  attention_kernel<<<dim3(NH, 64), 128, 0, stream>>>(qb, kb, vb, airN, woodN, leavesN, aT);
  proj_gemm_kernel<<<dim3(16, 64), 128, 0, stream>>>(wp, aT, proj_b, x, out);
}
